// GCNNet_18296560681308
// MI455X (gfx1250) — matrix operand order repaired and checked by reading, not compiled, not run on hardware
//
#include <hip/hip_runtime.h>
#include <hip/hip_bf16.h>
#include <stdint.h>

// ---------------------------------------------------------------------------
// Types for WMMA (CDNA5 / gfx1250, wave32)
// ---------------------------------------------------------------------------
typedef __bf16 bf16_t;
typedef bf16_t v16bf __attribute__((ext_vector_type(16)));
typedef float  v8f   __attribute__((ext_vector_type(8)));
typedef unsigned int v4u __attribute__((ext_vector_type(4)));

union ABCast { v4u u[2]; v16bf v; };   // 32 bytes: two b128 loads -> one WMMA operand

static __device__ __forceinline__ unsigned short f2bf(float x) {
  unsigned int u = __float_as_uint(x);
  u += 0x7FFFu + ((u >> 16) & 1u);     // round-to-nearest-even
  return (unsigned short)(u >> 16);
}

// ---------------------------------------------------------------------------
// Utility kernels
// ---------------------------------------------------------------------------
__global__ void k_fill_f32(float* __restrict__ p, float v, long long n) {
  long long i = (long long)blockIdx.x * blockDim.x + threadIdx.x;
  if (i < n) p[i] = v;
}

__global__ void k_deg_accum(const int* __restrict__ dst, float* __restrict__ deg, int E) {
  int e = blockIdx.x * blockDim.x + threadIdx.x;
  if (e < E) atomicAdd(&deg[dst[e]], 1.0f);
}

__global__ void k_deg_finish(float* __restrict__ dinv, int N) {
  int i = blockIdx.x * blockDim.x + threadIdx.x;
  if (i < N) dinv[i] = rsqrtf(dinv[i] + 1.0f);   // deg >= 1 -> finite
}

// x [N x K] f32 row-major  ->  out [N x Kpad] bf16, zero padded
__global__ void k_conv_rows_bf16(const float* __restrict__ x, unsigned short* __restrict__ out,
                                 int N, int K, int Kpad) {
  long long idx = (long long)blockIdx.x * blockDim.x + threadIdx.x;
  if (idx >= (long long)N * Kpad) return;
  int i = (int)(idx / Kpad), k = (int)(idx % Kpad);
  float v = (k < K) ? x[(long long)i * K + k] : 0.0f;
  out[idx] = f2bf(v);
}

// w [K x F] f32 row-major -> out [Fpad x Kpad] bf16 = transposed, zero padded.
// Zero rows f >= F guarantee the GEMM's padded output columns are exactly 0,
// which lets the GEMM store unconditionally into the padded C buffer.
__global__ void k_conv_wT_bf16(const float* __restrict__ w, unsigned short* __restrict__ out,
                               int K, int F, int Kpad, int Fpad) {
  int idx = blockIdx.x * blockDim.x + threadIdx.x;
  if (idx >= Fpad * Kpad) return;
  int f = idx / Kpad, k = idx % Kpad;
  float v = (k < K && f < F) ? w[k * F + f] : 0.0f;
  out[idx] = f2bf(v);
}

// ---------------------------------------------------------------------------
// WMMA GEMM: C[N x CSTRIDE] = A[N x KPAD](bf16) * BT[NCOLT*16 x KPAD](bf16)^T
// One wave -> one 16x16 tile; 8 waves/block sweep the column tiles.
// A fragments preloaded once per block row-tile (reused across col tiles);
// all B fragments for a tile preloaded as one clause so s_wait_loadcnt can
// drain gradually under the WMMA accumulation chain.
// Requires: gridDim.x * 16 == N (N multiple of 16) -> no row guards.
// Stores are unconditional: padded C columns receive exact zeros (B padded).
// ---------------------------------------------------------------------------
template<int KPAD, int NCOLT, int CSTRIDE>
__global__ void k_wmma_gemm(const unsigned short* __restrict__ A,
                            const unsigned short* __restrict__ BT,
                            float* __restrict__ C) {
  constexpr int NCHUNK = KPAD / 32;
  const int lane = threadIdx.x & 31;
  const int wave = threadIdx.x >> 5;
  const int m    = lane & 15;
  const int kh   = lane >> 4;                 // 0 or 1
  const int row0 = blockIdx.x * 16;

  // A (16-bit 16x32 layout): lane holds row m; kh selects K runs {kh*8..+7, 16+kh*8..+7}
  const unsigned short* aBase = A + (size_t)(row0 + m) * KPAD;
  ABCast a[NCHUNK];
#pragma unroll
  for (int c = 0; c < NCHUNK; ++c) {
    a[c].u[0] = *(const v4u*)(aBase + c * 32 + kh * 8);
    a[c].u[1] = *(const v4u*)(aBase + c * 32 + kh * 8 + 16);
  }

  for (int ct = wave; ct < NCOLT; ct += 8) {
    // B (16-bit 32x16 layout): lane holds col m; kh selects K runs {kh*8..+7, 16+kh*8..+7}
    const unsigned short* bBase = BT + (size_t)(ct * 16 + m) * KPAD;
    ABCast b[NCHUNK];
#pragma unroll
    for (int c = 0; c < NCHUNK; ++c) {
      b[c].u[0] = *(const v4u*)(bBase + c * 32 + kh * 8);
      b[c].u[1] = *(const v4u*)(bBase + c * 32 + kh * 8 + 16);
    }
    v8f acc = {};
#pragma unroll
    for (int c = 0; c < NCHUNK; ++c)
      acc = __builtin_amdgcn_wmma_f32_16x16x32_bf16(
              false, a[c].v, false, b[c].v, (short)0, acc, false, false);

    // C layout: n = lane%16, m = r + 8*kh
    float* cBase = C + (size_t)(row0 + kh * 8) * CSTRIDE + (ct * 16 + m);
#pragma unroll
    for (int r = 0; r < 8; ++r)
      cBase[(size_t)r * CSTRIDE] = acc[r];
  }
}

// ---------------------------------------------------------------------------
// Edge scatter: one wave per edge, lanes over features (L2-resident atomics)
// ---------------------------------------------------------------------------
__global__ void k_edge_scatter(const int* __restrict__ src, const int* __restrict__ dst,
                               const float* __restrict__ dinv,
                               const float* __restrict__ h, float* __restrict__ agg,
                               int E, int F, int stride) {
  long long gid = (long long)blockIdx.x * blockDim.x + threadIdx.x;
  int e = (int)(gid >> 5), lane = (int)(gid & 31);
  if (e >= E) return;
  int s = src[e], d = dst[e];
  float nrm = dinv[s] * dinv[d];
  const float* hr = h + (size_t)s * stride;
  float* ar = agg + (size_t)d * stride;
  for (int f = lane; f < F; f += 32)
    atomicAdd(ar + f, hr[f] * nrm);
}

// layer-1 epilogue: self-loop + bias + relu -> bf16 A-matrix for layer 2
__global__ void k_finish1(const float* __restrict__ agg, const float* __restrict__ h,
                          const float* __restrict__ dinv, const float* __restrict__ bias,
                          unsigned short* __restrict__ outbf,
                          int N, int F, int stride, int outStride) {
  long long idx = (long long)blockIdx.x * blockDim.x + threadIdx.x;
  if (idx >= (long long)N * outStride) return;
  int i = (int)(idx / outStride), f = (int)(idx % outStride);
  float v = 0.0f;
  if (f < F) {
    float di = dinv[i];
    v = fmaxf(agg[(size_t)i * stride + f] + h[(size_t)i * stride + f] * di * di + bias[f], 0.0f);
  }
  outbf[idx] = f2bf(v);
}

// layer-2 epilogue fused with segment_max pooling (relu'd vals >= 0, pool init 0;
// int-bit atomicMax is order-preserving for non-negative floats)
__global__ void k_finish2_pool(const float* __restrict__ agg, const float* __restrict__ h,
                               const float* __restrict__ dinv, const float* __restrict__ bias,
                               const int* __restrict__ batch, int* __restrict__ pooledBits,
                               int N, int F, int stride) {
  long long idx = (long long)blockIdx.x * blockDim.x + threadIdx.x;
  if (idx >= (long long)N * stride) return;
  int i = (int)(idx / stride), f = (int)(idx % stride);
  if (f >= F) return;
  float di = dinv[i];
  float v = fmaxf(agg[idx] + h[idx] * di * di + bias[f], 0.0f);
  atomicMax(&pooledBits[(size_t)batch[i] * stride + f], __float_as_int(v));
}

// ---------------------------------------------------------------------------
// Tiny graph-level MLP: one block per graph, LDS staged
// ---------------------------------------------------------------------------
__global__ void k_mlp(const float* __restrict__ pooled,
                      const float* __restrict__ Wg, const float* __restrict__ bg,
                      const float* __restrict__ Wf, const float* __restrict__ bf_,
                      const float* __restrict__ Wo, const float* __restrict__ bo,
                      float* __restrict__ out, int poolStride) {
  __shared__ float p[115];
  __shared__ float g1[64];
  __shared__ float g2[32];
  int g = blockIdx.x, t = threadIdx.x;
  for (int f = t; f < 115; f += 128) p[f] = pooled[(size_t)g * poolStride + f];
  __syncthreads();
  if (t < 64) {
    float s = bg[t];
    for (int k = 0; k < 115; ++k) s += p[k] * Wg[k * 64 + t];
    g1[t] = fmaxf(s, 0.0f);
  }
  __syncthreads();
  if (t < 32) {
    float s = bf_[t];
    for (int k = 0; k < 64; ++k) s += g1[k] * Wf[k * 32 + t];
    g2[t] = fmaxf(s, 0.0f);
  }
  __syncthreads();
  if (t == 0) {
    float s = bo[0];
    for (int k = 0; k < 32; ++k) s += g2[k] * Wo[k];
    out[g] = s;
  }
}

// ---------------------------------------------------------------------------
// Host side
// ---------------------------------------------------------------------------
extern "C" void kernel_launch(void* const* d_in, const int* in_sizes, int n_in,
                              void* d_out, int out_size, void* d_ws, size_t ws_size,
                              hipStream_t stream) {
  (void)in_sizes; (void)n_in; (void)out_size; (void)ws_size;

  constexpr int N  = 100000, E = 800000, G = 256;
  constexpr int K1 = 114, F1 = 230, K1P = 128, F1P = 240;   // layer 1
  constexpr int K2 = 230, F2 = 115, K2P = 256, F2P = 128;   // layer 2

  const float* x     = (const float*)d_in[0];
  const int*   ei    = (const int*)d_in[1];
  const int*   srcE  = ei;
  const int*   dstE  = ei + E;
  const int*   batch = (const int*)d_in[2];
  const float* W1 = (const float*)d_in[3];  const float* b1 = (const float*)d_in[4];
  const float* W2 = (const float*)d_in[5];  const float* b2 = (const float*)d_in[6];
  const float* Wg = (const float*)d_in[7];  const float* bg = (const float*)d_in[8];
  const float* Wf = (const float*)d_in[9];  const float* bff = (const float*)d_in[10];
  const float* Wo = (const float*)d_in[11]; const float* bo = (const float*)d_in[12];

  char* ws = (char*)d_ws;
  size_t off = 0;
  auto carve = [&](size_t bytes) { char* p = ws + off; off += (bytes + 511) & ~size_t(511); return p; };

  float*          dinv  = (float*)         carve((size_t)N * 4);
  unsigned short* Xbf   = (unsigned short*)carve((size_t)N * K1P * 2);
  unsigned short* W1T   = (unsigned short*)carve((size_t)F1P * K1P * 2);
  float*          h1    = (float*)         carve((size_t)N * F1P * 4);
  float*          agg1  = (float*)         carve((size_t)N * F1P * 4);
  unsigned short* A2bf  = (unsigned short*)carve((size_t)N * K2P * 2);
  unsigned short* W2T   = (unsigned short*)carve((size_t)F2P * K2P * 2);
  float*          h2    = (float*)         carve((size_t)N * F2P * 4);
  float*          agg2  = (float*)         carve((size_t)N * F2P * 4);
  int*            pool  = (int*)           carve((size_t)G * F2P * 4);

  const int B = 256;
  auto nb = [](long long n) { return dim3((unsigned)((n + 255) / 256)); };

  // degree -> dinv
  k_fill_f32  <<<nb(N), B, 0, stream>>>(dinv, 0.0f, N);
  k_deg_accum <<<nb(E), B, 0, stream>>>(dstE, dinv, E);
  k_deg_finish<<<nb(N), B, 0, stream>>>(dinv, N);

  // layer 1: convert, GEMM (WMMA bf16), scatter, epilogue
  k_conv_rows_bf16<<<nb((long long)N * K1P), B, 0, stream>>>(x, Xbf, N, K1, K1P);
  k_conv_wT_bf16  <<<nb(F1P * K1P), B, 0, stream>>>(W1, W1T, K1, F1, K1P, F1P);
  k_wmma_gemm<K1P, F1P / 16, F1P><<<dim3(N / 16), B, 0, stream>>>(Xbf, W1T, h1);
  k_fill_f32<<<nb((long long)N * F1P), B, 0, stream>>>(agg1, 0.0f, (long long)N * F1P);
  k_edge_scatter<<<nb((long long)E * 32), B, 0, stream>>>(srcE, dstE, dinv, h1, agg1, E, F1, F1P);
  k_finish1<<<nb((long long)N * K2P), B, 0, stream>>>(agg1, h1, dinv, b1, A2bf, N, F1, F1P, K2P);

  // layer 2: convert W, GEMM, scatter, epilogue fused with max-pool
  k_conv_wT_bf16<<<nb(F2P * K2P), B, 0, stream>>>(W2, W2T, K2, F2, K2P, F2P);
  k_wmma_gemm<K2P, F2P / 16, F2P><<<dim3(N / 16), B, 0, stream>>>(A2bf, W2T, h2);
  k_fill_f32<<<nb((long long)N * F2P), B, 0, stream>>>(agg2, 0.0f, (long long)N * F2P);
  k_edge_scatter<<<nb((long long)E * 32), B, 0, stream>>>(srcE, dstE, dinv, h2, agg2, E, F2, F2P);
  k_fill_f32<<<nb(G * F2P), B, 0, stream>>>((float*)pool, 0.0f, G * F2P);
  k_finish2_pool<<<nb((long long)N * F2P), B, 0, stream>>>(agg2, h2, dinv, b2, batch, pool, N, F2, F2P);

  // graph-level MLP -> d_out [256]
  k_mlp<<<dim3(G), 128, 0, stream>>>((const float*)pool, Wg, bg, Wf, bff, Wo, bo,
                                     (float*)d_out, F2P);
}
